// PoseMLP_60739427500697
// MI455X (gfx1250) — compile-verified, matrix operand order checked by reading, not run on hardware
//
#include <hip/hip_runtime.h>
#include <hip/hip_bf16.h>
#include <math.h>

typedef __attribute__((ext_vector_type(16))) _Float16 v16h;
typedef __attribute__((ext_vector_type(8)))  _Float16 v8h;
typedef __attribute__((ext_vector_type(8)))  float    v8f;

namespace {
constexpr int TPB_TOK = 128;   // tokens per block
constexpr int IN_DIM  = 45;
constexpr int K1      = 64;    // K of GEMM1, zero-padded from 45
constexpr int HDIM    = 128;
constexpr int DDIM    = 256;
constexpr int XS      = 72;    // sX row stride (halves); 144B, 16B-aligned
constexpr int WS      = 72;    // sW1t row stride (halves)
constexpr int HS      = 136;   // sH row stride (halves); 272B, 16B-aligned
constexpr int W2S     = 136;   // sW2t row stride (halves)
}

// Branch-free exact-GELU: erf via Abramowitz-Stegun 7.1.26 (|err| <= 1.5e-7),
// one v_rcp + one v_exp + fma chain, no EXEC divergence.
__device__ __forceinline__ float gelu_erf(float x) {
    const float z  = x * 0.70710678118654752f;
    const float az = __builtin_fabsf(z);
    const float t  = __builtin_amdgcn_rcpf(__builtin_fmaf(0.3275911f, az, 1.0f));
    float p = __builtin_fmaf(1.061405429f, t, -1.453152027f);
    p = __builtin_fmaf(p, t, 1.421413741f);
    p = __builtin_fmaf(p, t, -0.284496736f);
    p = __builtin_fmaf(p, t, 0.254829592f);
    p = p * t;
    const float e      = __expf(-az * az);
    const float erf_az = __builtin_fmaf(-p, e, 1.0f);
    const float erfz   = __builtin_copysignf(erf_az, z);
    return 0.5f * x * (1.0f + erfz);
}

// Build a 16-half WMMA operand from K-major LDS: two 16B loads + shuffle.
__device__ __forceinline__ v16h frag_from_kmajor(const _Float16* base) {
    const v8h lo = *(const v8h*)(base);        // K = kgrp .. kgrp+7
    const v8h hi = *(const v8h*)(base + 16);   // K = kgrp+16 .. kgrp+23
    return __builtin_shufflevector(lo, hi, 0, 1, 2, 3, 4, 5, 6, 7,
                                           8, 9, 10, 11, 12, 13, 14, 15);
}

__global__ __launch_bounds__(256)
void pose_mlp_fused(const float* __restrict__ pose,
                    const float* __restrict__ ln1_g, const float* __restrict__ ln1_b,
                    const float* __restrict__ w1,    const float* __restrict__ b1,
                    const float* __restrict__ w2,    const float* __restrict__ b2,
                    const float* __restrict__ ln2_g, const float* __restrict__ ln2_b,
                    float* __restrict__ out)
{
    __shared__ _Float16 sX  [TPB_TOK * XS];   // [row][k]   LN1 output, K-major
    __shared__ _Float16 sW1t[HDIM * WS];      // [n][k]     w1 transposed, K-major
    __shared__ _Float16 sH  [TPB_TOK * HS];   // [row][h]   GELU output, K-major
    __shared__ _Float16 sW2t[DDIM * W2S];     // [d][k]     w2 transposed, K-major

    const int tid  = threadIdx.x;
    const int lane = tid & 31;
    const int wave = tid >> 5;                // 8 waves; wave == M-tile index
    const int hi   = lane >> 4;               // half-wave (0/1)
    const int ln16 = lane & 15;
    const int blk  = blockIdx.x;
    const int kgrp = hi * 8;                  // per-lane K group base

    // ---- stage w1^T (fp32->f16, zero-pad K 45..63); coalesced global reads ----
    for (int i = tid; i < HDIM * K1; i += 256) {
        const int n = i & (HDIM - 1);         // consecutive lanes -> consecutive n
        const int k = i >> 7;
        const float v = (k < IN_DIM) ? w1[k * HDIM + n] : 0.0f;
        sW1t[n * WS + k] = (_Float16)v;
    }
    // ---- stage w2^T (fp32->f16); coalesced global reads ----
    for (int i = tid; i < DDIM * HDIM; i += 256) {
        const int d = i & (DDIM - 1);         // consecutive lanes -> consecutive d
        const int k = i >> 8;
        sW2t[d * W2S + k] = (_Float16)w2[k * DDIM + d];
    }

    // ---- LN1: one thread per token row (45 features) ----
    if (tid < TPB_TOK) {
        const long token = (long)blk * TPB_TOK + tid;
        const float* p = pose + token * IN_DIM;
        float x[IN_DIM];
        float s = 0.0f;
        #pragma unroll
        for (int j = 0; j < IN_DIM; ++j) { x[j] = p[j]; s += x[j]; }
        const float mu = s * (1.0f / IN_DIM);
        float vs = 0.0f;
        #pragma unroll
        for (int j = 0; j < IN_DIM; ++j) { const float d = x[j] - mu; vs += d * d; }
        const float rstd = rsqrtf(vs * (1.0f / IN_DIM) + 1e-5f);
        #pragma unroll
        for (int j = 0; j < IN_DIM; ++j)
            sX[tid * XS + j] = (_Float16)((x[j] - mu) * rstd * ln1_g[j] + ln1_b[j]);
        #pragma unroll
        for (int j = IN_DIM; j < K1; ++j)
            sX[tid * XS + j] = (_Float16)0.0f;
    }
    __syncthreads();

    // ---- GEMM1: [128x64] x [64x128] -> +b1 -> GELU -> sH (f16) ----
    v16h a1[2];
    #pragma unroll
    for (int ks = 0; ks < 2; ++ks)
        a1[ks] = frag_from_kmajor(&sX[(wave * 16 + ln16) * XS + ks * 32 + kgrp]);

    #pragma unroll
    for (int n = 0; n < 8; ++n) {
        v8f c = {};
        #pragma unroll
        for (int ks = 0; ks < 2; ++ks) {
            const v16h bf =
                frag_from_kmajor(&sW1t[(n * 16 + ln16) * WS + ks * 32 + kgrp]);
            c = __builtin_amdgcn_wmma_f32_16x16x32_f16(false, a1[ks], false, bf,
                                                       (short)0, c, false, false);
        }
        const float b1v = b1[n * 16 + ln16];
        #pragma unroll
        for (int r = 0; r < 8; ++r) {
            const float g = gelu_erf(c[r] + b1v);
            sH[(wave * 16 + r + hi * 8) * HS + n * 16 + ln16] = (_Float16)g;
        }
    }
    __syncthreads();

    // ---- GEMM2: [128x128] x [128x256] -> +b2, stats accumulated in-flight ----
    v16h a2[4];
    #pragma unroll
    for (int ks = 0; ks < 4; ++ks)
        a2[ks] = frag_from_kmajor(&sH[(wave * 16 + ln16) * HS + ks * 32 + kgrp]);

    v8f  acc[16];
    float lsum[8], lsq[8];
    #pragma unroll
    for (int r = 0; r < 8; ++r) { lsum[r] = 0.0f; lsq[r] = 0.0f; }

    #pragma unroll
    for (int n = 0; n < 16; ++n) {
        v8f c = {};
        #pragma unroll
        for (int ks = 0; ks < 4; ++ks) {
            const v16h bf =
                frag_from_kmajor(&sW2t[(n * 16 + ln16) * W2S + ks * 32 + kgrp]);
            c = __builtin_amdgcn_wmma_f32_16x16x32_f16(false, a2[ks], false, bf,
                                                       (short)0, c, false, false);
        }
        const float b2v = b2[n * 16 + ln16];
        #pragma unroll
        for (int r = 0; r < 8; ++r) {
            const float v = c[r] + b2v;
            c[r] = v;
            lsum[r] += v;
            lsq[r]  += v * v;
        }
        acc[n] = c;
    }

    // ---- LN2: reduce across the 16 lanes of each half-wave ----
    #pragma unroll
    for (int r = 0; r < 8; ++r) {
        float s = lsum[r], q = lsq[r];
        #pragma unroll
        for (int mask = 1; mask < 16; mask <<= 1) {
            s += __shfl_xor(s, mask, 32);
            q += __shfl_xor(q, mask, 32);
        }
        lsum[r] = s; lsq[r] = q;
    }

    #pragma unroll
    for (int r = 0; r < 8; ++r) {
        const float mu   = lsum[r] * (1.0f / DDIM);
        const float var  = lsq[r] * (1.0f / DDIM) - mu * mu;
        const float rstd = rsqrtf(var + 1e-5f);
        const long token = (long)blk * TPB_TOK + wave * 16 + r + hi * 8;
        #pragma unroll
        for (int n = 0; n < 16; ++n) {
            const int col = n * 16 + ln16;
            out[token * DDIM + col] = (acc[n][r] - mu) * rstd * ln2_g[col] + ln2_b[col];
        }
    }
}

extern "C" void kernel_launch(void* const* d_in, const int* in_sizes, int n_in,
                              void* d_out, int out_size, void* d_ws, size_t ws_size,
                              hipStream_t stream) {
    (void)in_sizes; (void)n_in; (void)out_size; (void)d_ws; (void)ws_size;
    const float* pose  = (const float*)d_in[0];
    const float* ln1_g = (const float*)d_in[1];
    const float* ln1_b = (const float*)d_in[2];
    const float* w1    = (const float*)d_in[3];
    const float* b1    = (const float*)d_in[4];
    const float* w2    = (const float*)d_in[5];
    const float* b2    = (const float*)d_in[6];
    const float* ln2_g = (const float*)d_in[7];
    const float* ln2_b = (const float*)d_in[8];
    float* out = (float*)d_out;

    const int tokens = 64 * 4096;
    dim3 grid(tokens / TPB_TOK), block(256);
    pose_mlp_fused<<<grid, block, 0, stream>>>(pose, ln1_g, ln1_b, w1, b1,
                                               w2, b2, ln2_g, ln2_b, out);
}
